// NTXentLoss_73615739453601
// MI455X (gfx1250) — matrix operand order repaired and checked by reading, not compiled, not run on hardware
//
#include <hip/hip_runtime.h>
#include <hip/hip_fp16.h>

typedef __attribute__((ext_vector_type(16))) _Float16 v16h;
typedef __attribute__((ext_vector_type(8)))  _Float16 v8h;
typedef __attribute__((ext_vector_type(4)))  _Float16 v4h;
typedef __attribute__((ext_vector_type(8)))  float    v8f;
typedef __attribute__((ext_vector_type(4)))  float    v4f;

typedef __attribute__((ext_vector_type(4)))  unsigned int u32x4;
typedef __attribute__((ext_vector_type(8)))  int          i32x8;
typedef __attribute__((ext_vector_type(4)))  int          i32x4;

#define N_BATCH 128
#define B_HALF  256
#define TWO_B   512
#define DDIM    128
#define INV_T   2.0f        // 1 / TEMPERATURE
#define SHIFT   2.0f        // max possible scaled similarity (|cos|<=1, /0.5)
#define LOG2E   1.4426950408889634f
#define LDS_BYTES (TWO_B * DDIM * 2)   // 512*128 f16 = 128 KB (CDNA5: 320KB/WGP)

// Raw v_exp_f32: argument range is [-5.8, 0] here, so the libm subnormal
// range-fixup (v_cmp/v_cndmask/v_ldexp) is provably dead weight.
__device__ __forceinline__ float fast_exp2(float x) {
#if __has_builtin(__builtin_amdgcn_exp2f)
    return __builtin_amdgcn_exp2f(x);
#else
    return exp2f(x);
#endif
}

// ---------------- Phase 1: L2-normalize rows, f32 -> f16 workspace ----------
__global__ __launch_bounds__(256) void ntxent_normalize(
    const float* __restrict__ zis, const float* __restrict__ zjs,
    _Float16* __restrict__ xf16, float* __restrict__ out)
{
    if (blockIdx.x == 0 && threadIdx.x == 0) out[0] = 0.0f;  // zero scalar output

    const int wave = threadIdx.x >> 5;
    const int lane = threadIdx.x & 31;
    const int row  = blockIdx.x * 8 + wave;            // 0 .. 65535
    const int n    = row >> 9;                         // batch
    const int k    = row & (TWO_B - 1);                // position in [zjs ; zis]

    const float* src = (k < B_HALF)
        ? (zjs + ((size_t)n * B_HALF + k) * DDIM)
        : (zis + ((size_t)n * B_HALF + (k - B_HALF)) * DDIM);

    v4f v = *(const v4f*)(src + lane * 4);
    float ss = v.x * v.x + v.y * v.y + v.z * v.z + v.w * v.w;
#pragma unroll
    for (int m = 16; m >= 1; m >>= 1) ss += __shfl_xor(ss, m, 32);

    const float inv = 1.0f / fmaxf(sqrtf(ss), 1e-8f);

    v4h o;
    o[0] = (_Float16)(v.x * inv);
    o[1] = (_Float16)(v.y * inv);
    o[2] = (_Float16)(v.z * inv);
    o[3] = (_Float16)(v.w * inv);
    *(v4h*)(xf16 + (size_t)row * DDIM + lane * 4) = o;
}

// ---------------- Phase 2: fused  X·Xᵀ (WMMA f16)  +  masked LSE  -----------
// One block per batch (1024 threads = 32 waves). Wave w owns M-tile w
// (rows 16w..16w+15) and sweeps all 32 N-tiles, K = 4 x 32.
__global__ __launch_bounds__(1024) void ntxent_sim_lse(
    const _Float16* __restrict__ xf16, float* __restrict__ out)
{
    extern __shared__ _Float16 lds[];                  // 512 x 128 f16, row-major

    const int n    = blockIdx.x;
    const int tid  = threadIdx.x;
    // wave id is uniform within a wave -> force to SGPR so tile-selection
    // branches below are scalar (s_cmp/s_cbranch), not v_cndmask chains.
    const int waveU = __builtin_amdgcn_readfirstlane(tid >> 5);   // M-tile 0..31
    const int lane = tid & 31;
    const int half = lane >> 4;                        // 0: lanes 0-15, 1: 16-31
    const int lcol = lane & 15;                        // N-column within tile

    // ---- stage whole batch operand into LDS via the Tensor Data Mover -----
#if __has_builtin(__builtin_amdgcn_tensor_load_to_lds)
    if (waveU == 0) {
        // Tensor DMA descriptor (CDNA5 ISA):
        // 2D tensor, 512 rows x 128 cols, 2-byte elements, tile == tensor.
        const unsigned long long ga =
            (unsigned long long)(const void*)(xf16 + (size_t)n * TWO_B * DDIM);
        u32x4 g0;
        g0[0] = 1u;                                    // count=1, user mode
        g0[1] = 0u;                                    // lds_addr = 0 (dyn-LDS base)
        g0[2] = (unsigned)(ga & 0xFFFFFFFFu);          // global_addr[31:0]
        g0[3] = (unsigned)((ga >> 32) & 0x01FFFFFFu)   // global_addr[56:32]
              | (2u << 30);                            // type = 2 ("image")
        i32x8 g1;
        g1[0] = 0x00010000;                            // wg_mask=0, data_size=1 (2B)
        g1[1] = (int)((DDIM & 0xFFFF) << 16);          // tensor_dim0[15:0] @ bit48
        g1[2] = (int)((TWO_B & 0xFFFF) << 16);         // tensor_dim1[15:0] @ bit80
        g1[3] = (int)((DDIM & 0xFFFF) << 16);          // tile_dim0 @ bit112
        g1[4] = TWO_B;                                 // tile_dim1 @ bit128 (tile_dim2=0)
        g1[5] = DDIM;                                  // tensor_dim0_stride @ bit160
        g1[6] = 0;
        g1[7] = (int)((TWO_B * (unsigned long long)DDIM) >> 16); // dim1_stride @ bit208
        i32x4 gz = {0, 0, 0, 0};                       // groups 2/3: dims unused (2D)
#if __clang_major__ >= 23
        i32x8 gz8 = {0, 0, 0, 0, 0, 0, 0, 0};
        __builtin_amdgcn_tensor_load_to_lds(g0, g1, gz, gz, gz8, 0);
#else
        __builtin_amdgcn_tensor_load_to_lds(g0, g1, gz, gz, 0);
#endif
#if __has_builtin(__builtin_amdgcn_s_wait_tensorcnt)
        __builtin_amdgcn_s_wait_tensorcnt(0);
#else
        asm volatile("s_wait_tensorcnt 0x0" ::: "memory");
#endif
    }
#else
    {
        const uint4* g = (const uint4*)(xf16 + (size_t)n * TWO_B * DDIM);
        uint4*       s = (uint4*)lds;
#pragma unroll
        for (int i = 0; i < 8; ++i) s[tid + i * 1024] = g[tid + i * 1024];
    }
#endif
    __syncthreads();

    // ---- A fragments for this wave's 16 rows (ISA 16-bit 16x32 layout) ----
    // lane<16: elems 0-7 = K+0..7, 8-15 = K+16..23 ; lane>=16: +8 on both.
    const int arow = 16 * waveU + lcol;
    v16h afrag[4];
#pragma unroll
    for (int ks = 0; ks < 4; ++ks) {
        const int k0 = ks * 32 + half * 8;
        v8h lo = *(const v8h*)(lds + arow * DDIM + k0);
        v8h hi = *(const v8h*)(lds + arow * DDIM + k0 + 16);
#pragma unroll
        for (int e = 0; e < 8; ++e) { afrag[ks][e] = lo[e]; afrag[ks][e + 8] = hi[e]; }
    }

    float sred[8];                                     // per-lane exp-sums, row slot v
#pragma unroll
    for (int v = 0; v < 8; ++v) sred[v] = 0.0f;
    float pos_acc = 0.0f;
    const int posTile = (waveU + 16) & 31;             // (row+B) % 2B tile

#pragma unroll 2
    for (int jt = 0; jt < 32; ++jt) {
        // B column N = X row (sim = X·Xᵀ). B layout matches A: elems 0-7 = K+0..7,
        // 8-15 = K+16..23; lanes 16-31 take +8 on both.
        const _Float16* bp = lds + (16 * jt + lcol) * DDIM + half * 8;
        v16h b[4];
#pragma unroll
        for (int ks = 0; ks < 4; ++ks) {
            v8h lo = *(const v8h*)(bp + ks * 32);
            v8h hi = *(const v8h*)(bp + ks * 32 + 16);
#pragma unroll
            for (int e = 0; e < 8; ++e) { b[ks][e] = lo[e]; b[ks][e + 8] = hi[e]; }
        }
        v8f c = {};
#pragma unroll
        for (int ks = 0; ks < 4; ++ks)
            c = __builtin_amdgcn_wmma_f32_16x16x32_f16(
                    false, afrag[ks], false, b[ks], (short)0, c, false, false);

        // C layout: c[v] = sim(row 16w + v + 8*half, col 16*jt + lcol)
        // Hot path: 1 v_fma + 1 v_exp + 1 v_add per element (exact fixed-shift LSE).
#pragma unroll
        for (int v = 0; v < 8; ++v)
            sred[v] += fast_exp2(fmaf(c[v], INV_T * LOG2E, -SHIFT * LOG2E));

        // Rare, wave-uniform corrections (scalar branch, taken once each):
        if (jt == waveU) {                             // remove self-similarity
#pragma unroll
            for (int v = 0; v < 8; ++v)
                if (lcol == v + 8 * half)
                    sred[v] -= fast_exp2(fmaf(c[v], INV_T * LOG2E, -SHIFT * LOG2E));
        }
        if (jt == posTile) {                           // grab positive-pair logit
#pragma unroll
            for (int v = 0; v < 8; ++v)
                if (lcol == v + 8 * half)
                    pos_acc += c[v] * INV_T;
        }
    }

    // ---- reduce exp-sums across the 16 lanes holding each row's columns ----
#pragma unroll
    for (int v = 0; v < 8; ++v) {
        float s = sred[v];
#pragma unroll
        for (int m = 1; m <= 8; m <<= 1) s += __shfl_xor(s, m, 32);
        sred[v] = s;
    }
    // positive logits: each row's pos lives on exactly one lane -> full-wave sum
#pragma unroll
    for (int m = 1; m <= 16; m <<= 1) pos_acc += __shfl_xor(pos_acc, m, 32);

    float lsesum = 0.0f;                               // rows 0-7 (half0) / 8-15 (half1)
#pragma unroll
    for (int v = 0; v < 8; ++v) lsesum += SHIFT + __logf(sred[v]);
    lsesum += __shfl_xor(lsesum, 16, 32);              // combine halves -> 16-row total

    if (lane == 0)
        atomicAdd(out, (lsesum - pos_acc) * (1.0f / (float)TWO_B));
}

extern "C" void kernel_launch(void* const* d_in, const int* in_sizes, int n_in,
                              void* d_out, int out_size, void* d_ws, size_t ws_size,
                              hipStream_t stream) {
    const float* zis = (const float*)d_in[0];
    const float* zjs = (const float*)d_in[1];
    float* out       = (float*)d_out;
    _Float16* xf16   = (_Float16*)d_ws;   // 65536 rows x 128 f16 = 16 MB scratch

    // allow 128 KB dynamic LDS (CDNA5 WGP has 320 KB)
    hipFuncSetAttribute((const void*)ntxent_sim_lse,
                        hipFuncAttributeMaxDynamicSharedMemorySize, LDS_BYTES);

    ntxent_normalize<<<8192, 256, 0, stream>>>(zis, zjs, xf16, out);
    ntxent_sim_lse<<<N_BATCH, 1024, LDS_BYTES, stream>>>(xf16, out);
}
